// MHA_13237089206867
// MI455X (gfx1250) — matrix operand order repaired and checked by reading, not compiled, not run on hardware
//
#include <hip/hip_runtime.h>

// ---------------------------------------------------------------------------
// MHA forward for MI455X (gfx1250): bf16 WMMA + TDM (tensor_load_to_lds)
// staged GEMMs AND attention K/V staging, flash attention with LDS P-transpose.
// ---------------------------------------------------------------------------

typedef __bf16 bf16;
typedef __attribute__((ext_vector_type(16))) __bf16 v16bf;
typedef __attribute__((ext_vector_type(8)))  float  v8f;
typedef __attribute__((ext_vector_type(4)))  unsigned v4u;
typedef __attribute__((ext_vector_type(8)))  int      v8i;
typedef __attribute__((ext_vector_type(4)))  int      v4i;

constexpr int B_  = 4;
constexpr int L_  = 2048;
constexpr int D_  = 2048;
constexpr int H_  = 16;
constexpr int HD_ = 128;
constexpr float ATTN_SCALE = 0.08838834764831845f;   // 1/sqrt(128)

__device__ __forceinline__ v8f wmma_bf16(v16bf a, v16bf b, v8f c) {
  return __builtin_amdgcn_wmma_f32_16x16x32_bf16(
      false, a, false, b, (short)0, c, false, false);
}

// ---------------------------------------------------------------------------
// Fragment loaders (ISA 7.12.2 layouts).
// ---------------------------------------------------------------------------
__device__ __forceinline__ v16bf load_a_frag_g(const bf16* __restrict__ base,
                                               int ldk, int kbase, int lane) {
  int ml = lane & 15, half = lane >> 4;
  const bf16* p = base + (size_t)ml * ldk + kbase + half * 8;
  union { uint4 u[2]; v16bf v; } f;
  f.u[0] = *(const uint4*)(p);
  f.u[1] = *(const uint4*)(p + 16);
  return f.v;
}
// LDS tile variants; byte row-stride chosen for bank-conflict-free b128 reads.
__device__ __forceinline__ v16bf lds_a_frag(const char* base, int rstride,
                                            int kbase, int lane) {
  int ml = lane & 15, half = lane >> 4;
  const char* p = base + ml * rstride + (kbase + half * 8) * 2;
  union { uint4 u[2]; v16bf v; } f;
  f.u[0] = *(const uint4*)(p);
  f.u[1] = *(const uint4*)(p + 32);
  return f.v;
}
__device__ __forceinline__ v16bf lds_b_frag(const char* base, int rstride,
                                            int kbase, int lane) {
  int nl = lane & 15, kg = lane >> 4;
  const char* p = base + nl * rstride + (kbase + kg * 8) * 2;
  union { uint4 u[2]; v16bf v; } f;
  f.u[0] = *(const uint4*)(p);
  f.u[1] = *(const uint4*)(p + 32);
  return f.v;
}

// ---------------------------------------------------------------------------
// TDM: 2-D tensor_load_to_lds of a (tile_d0 x tile_d1) bf16 tile.
// D# packing per CDNA5 ISA 8.3/8.4 with LDS row padding.
// ---------------------------------------------------------------------------
#if __has_include(<hip/amd_detail/amd_gfx1250_TDM.h>)
#define TDM_ARITY6 1
#else
#define TDM_ARITY6 0
#endif

__device__ __forceinline__ void tdm_load_tile(unsigned lds_off, const void* gptr,
                                              unsigned tile_d0, unsigned tile_d1,
                                              unsigned stride_elems,
                                              unsigned tensor_d0,
                                              int pad_interval, int pad_amount) {
  unsigned long long ga = (unsigned long long)gptr;
  v4u g0;
  g0[0] = 1u;                                   // count=1, user descriptor
  g0[1] = lds_off;                              // lds_addr (bytes)
  g0[2] = (unsigned)(ga & 0xffffffffu);         // global_addr[31:0]
  g0[3] = (unsigned)((ga >> 32) & 0x01ffffffu)  // global_addr[56:32]
          | (2u << 30);                         // type=2 ("image")
  v8i g1;
  g1[0] = (1 << 16)                             // data_size = 2 bytes
        | (1 << 20)                             // pad_enable
        | (pad_interval << 22)                  // dwords-per-pad code
        | (pad_amount << 25);                   // pad dwords code
  g1[1] = (int)((tensor_d0 & 0xffffu) << 16);   // tensor_dim0[15:0]
  g1[2] = (int)((tensor_d0 >> 16) & 0xffffu);   // tensor_dim0[31:16], dim1 lo=0
  g1[3] = 16 | (int)(tile_d0 << 16);            // tensor_dim1 hi (=1M), tile_dim0
  g1[4] = (int)tile_d1;                         // tile_dim1, tile_dim2=0
  g1[5] = (int)stride_elems;                    // tensor_dim0_stride[31:0]
  g1[6] = 0;
  g1[7] = 0;
  v4i z4 = (v4i){0, 0, 0, 0};
#if TDM_ARITY6
  v8i z8 = (v8i){0, 0, 0, 0, 0, 0, 0, 0};
  __builtin_amdgcn_tensor_load_to_lds(g0, g1, z4, z4, z8, 0);
#else
  __builtin_amdgcn_tensor_load_to_lds(g0, g1, z4, z4, 0);
#endif
}

// ---------------------------------------------------------------------------
// Kernel: fp32 -> bf16 conversion
// ---------------------------------------------------------------------------
__global__ void cvt_f32_bf16(const float* __restrict__ in,
                             bf16* __restrict__ out, size_t n) {
  size_t i = (size_t)blockIdx.x * blockDim.x + threadIdx.x;
  if (i < n) out[i] = (bf16)in[i];
}

// ---------------------------------------------------------------------------
// Kernel: C(MxN fp32) = A(MxK bf16 row-major) * Bw(NxK bf16 row-major)^T
// 8 waves, block tile 128x128, K slab 64, TDM double-buffered LDS.
// ---------------------------------------------------------------------------
constexpr int TILE_ROWSTRIDE = 144;                     // 64*2 + 16B pad
constexpr int TILE_BYTES     = 128 * TILE_ROWSTRIDE;    // 18432

template <int N, int K>
__global__ __launch_bounds__(256)
void gemm_tdm_kernel(const bf16* __restrict__ A, const bf16* __restrict__ Bw,
                     float* __restrict__ C) {
  const int lane = threadIdx.x & 31;
  const int wave = threadIdx.x >> 5;
  const int m0w = (wave & 3) * 32;
  const int n0w = (wave >> 2) * 64;
  const size_t m0 = (size_t)blockIdx.y * 128;
  const size_t n0 = (size_t)blockIdx.x * 128;

  __shared__ __align__(16) char ltile[2][2][TILE_BYTES];   // [buf][A=0/B=1]

  v8f acc[2][4];
#pragma unroll
  for (int i = 0; i < 2; ++i)
#pragma unroll
    for (int j = 0; j < 4; ++j)
      acc[i][j] = (v8f){0.f, 0.f, 0.f, 0.f, 0.f, 0.f, 0.f, 0.f};

  unsigned offA[2], offB[2];   // generic->32-bit trunc == LDS byte offset
#pragma unroll
  for (int b = 0; b < 2; ++b) {
    offA[b] = (unsigned)(unsigned long long)(void*)&ltile[b][0][0];
    offB[b] = (unsigned)(unsigned long long)(void*)&ltile[b][1][0];
  }

  constexpr int kIters = K / 64;

  if (wave == 0) {
    tdm_load_tile(offA[0], A + m0 * K, 64, 128, K, K, 4, 3);
    tdm_load_tile(offB[0], Bw + n0 * K, 64, 128, K, K, 4, 3);
  }
  __builtin_amdgcn_s_wait_tensorcnt(0);
  __syncthreads();

  for (int it = 0; it < kIters; ++it) {
    const int buf = it & 1;
    if (wave == 0 && it + 1 < kIters) {
      const int knext = (it + 1) * 64;
      tdm_load_tile(offA[buf ^ 1], A + m0 * K + knext, 64, 128, K, K, 4, 3);
      tdm_load_tile(offB[buf ^ 1], Bw + n0 * K + knext, 64, 128, K, K, 4, 3);
    }

    const char* aB = &ltile[buf][0][0] + m0w * TILE_ROWSTRIDE;
    const char* bB = &ltile[buf][1][0] + n0w * TILE_ROWSTRIDE;
#pragma unroll
    for (int kk = 0; kk < 64; kk += 32) {
      v16bf a0 = lds_a_frag(aB, TILE_ROWSTRIDE, kk, lane);
      v16bf a1 = lds_a_frag(aB + 16 * TILE_ROWSTRIDE, TILE_ROWSTRIDE, kk, lane);
      v16bf bfr[4];
#pragma unroll
      for (int j = 0; j < 4; ++j)
        bfr[j] = lds_b_frag(bB + j * 16 * TILE_ROWSTRIDE, TILE_ROWSTRIDE, kk, lane);
#pragma unroll
      for (int j = 0; j < 4; ++j) {
        acc[0][j] = wmma_bf16(a0, bfr[j], acc[0][j]);
        acc[1][j] = wmma_bf16(a1, bfr[j], acc[1][j]);
      }
    }

    __builtin_amdgcn_s_wait_tensorcnt(0);
    __syncthreads();
  }

  const int nl = lane & 15, mh = (lane >> 4) * 8;
#pragma unroll
  for (int i = 0; i < 2; ++i)
#pragma unroll
    for (int j = 0; j < 4; ++j) {
      float* cp = C + (m0 + m0w + i * 16 + mh) * N + n0 + n0w + j * 16 + nl;
#pragma unroll
      for (int v = 0; v < 8; ++v) cp[(size_t)v * N] = acc[i][j][v];
    }
}

// ---------------------------------------------------------------------------
// Kernel: RoPE on q,k + split/layout.  qkv fp32 (B,L,3D) ->
//   Qb,Kb bf16 (B,H,L,HD)  and  Vt bf16 (B,H,HD,L)  (V transposed).
// ---------------------------------------------------------------------------
__global__ void rope_split_kernel(const float* __restrict__ qkv,
                                  bf16* __restrict__ Qb, bf16* __restrict__ Kb,
                                  bf16* __restrict__ Vt) {
  const int HP = HD_ / 2;
  size_t idx = (size_t)blockIdx.x * blockDim.x + threadIdx.x;
  size_t total = (size_t)B_ * L_ * H_ * HP;
  if (idx >= total) return;
  int i = idx % HP;
  int h = (idx / HP) % H_;
  int l = (idx / ((size_t)HP * H_)) % L_;
  int b = idx / ((size_t)HP * H_ * L_);

  float inv_freq = __expf(-(float)i * (9.210340371976184f / (float)HP));
  float ang = (float)l * inv_freq;
  float c = cosf(ang), s = sinf(ang);

  const float* row = qkv + ((size_t)b * L_ + l) * (3 * D_);
  int e = h * HD_ + 2 * i;

  float q1 = row[e],          q2 = row[e + 1];
  float k1 = row[D_ + e],     k2 = row[D_ + e + 1];
  float v1 = row[2 * D_ + e], v2 = row[2 * D_ + e + 1];

  size_t qk = (((size_t)b * H_ + h) * L_ + l) * HD_ + 2 * i;
  Qb[qk]     = (bf16)(q1 * c - q2 * s);
  Qb[qk + 1] = (bf16)(q2 * c + q1 * s);
  Kb[qk]     = (bf16)(k1 * c - k2 * s);
  Kb[qk + 1] = (bf16)(k2 * c + k1 * s);

  size_t vb = ((size_t)b * H_ + h) * (size_t)HD_ * L_;
  Vt[vb + (size_t)(2 * i)     * L_ + l] = (bf16)v1;
  Vt[vb + (size_t)(2 * i + 1) * L_ + l] = (bf16)v2;
}

// ---------------------------------------------------------------------------
// Kernel: causal flash attention, block-cooperative K/V staging via TDM.
// Block = 4 waves on 4 consecutive 16-query tiles of one (b,h); each 32-key
// slab of K (32x128) and V^T (128x32) is TDM-loaded once into LDS and shared.
// ---------------------------------------------------------------------------
constexpr int KT_STRIDE = 272;   // 128*2 + 16B pad (conflict-free b128)
constexpr int VT_STRIDE = 80;    //  32*2 + 16B pad (conflict-free b128)

__global__ __launch_bounds__(128)
void attn_kernel(const bf16* __restrict__ Qb, const bf16* __restrict__ Kb,
                 const bf16* __restrict__ Vt, bf16* __restrict__ Yb) {
  const int lane = threadIdx.x & 31;
  const int wave = threadIdx.x >> 5;
  const int QB = L_ / 64;                 // 64-query super-tiles per (b,h)
  const int qBlk = blockIdx.x % QB;
  const int bh   = blockIdx.x / QB;
  const int q0   = qBlk * 64 + wave * 16; // this wave's 16-query tile
  const int h = bh % H_, b = bh / H_;

  const bf16* Qrow  = Qb + ((size_t)bh * L_ + q0) * HD_;
  const bf16* Kbase = Kb + (size_t)bh * L_ * HD_;
  const bf16* Vbase = Vt + (size_t)bh * HD_ * L_;

  __shared__ __align__(16) char ldsK[2][32 * KT_STRIDE];    // 2 x 8704 B
  __shared__ __align__(16) char ldsV[2][128 * VT_STRIDE];   // 2 x 10240 B
  __shared__ __align__(16) bf16 ldsP[4][16 * 32];           // per-wave P tile
  bf16* Pl = ldsP[wave];

  unsigned offK[2], offV[2];
#pragma unroll
  for (int i = 0; i < 2; ++i) {
    offK[i] = (unsigned)(unsigned long long)(void*)&ldsK[i][0];
    offV[i] = (unsigned)(unsigned long long)(void*)&ldsV[i][0];
  }

  const int nl = lane & 15, half = lane >> 4;

  v16bf qf[4];
#pragma unroll
  for (int kk = 0; kk < 4; ++kk) qf[kk] = load_a_frag_g(Qrow, HD_, kk * 32, lane);

  v8f o[8];
#pragma unroll
  for (int t = 0; t < 8; ++t) o[t] = (v8f){0.f,0.f,0.f,0.f,0.f,0.f,0.f,0.f};
  float mrow[8], lrow[8];
#pragma unroll
  for (int v = 0; v < 8; ++v) { mrow[v] = -__builtin_inff(); lrow[v] = 0.f; }

  const int nIter = 2 * qBlk + 2;         // 32-key slabs up to qBlk*64+63

  if (wave == 0) {
    tdm_load_tile(offK[0], Kbase, 128, 32, HD_, HD_, 5, 3);
    tdm_load_tile(offV[0], Vbase, 32, 128, L_, L_, 3, 3);
  }
  __builtin_amdgcn_s_wait_tensorcnt(0);
  __syncthreads();

  for (int it = 0; it < nIter; ++it) {
    const int buf = it & 1;
    const int kb = it * 32;
    if (wave == 0 && it + 1 < nIter) {
      const int kn = kb + 32;
      tdm_load_tile(offK[buf ^ 1], Kbase + (size_t)kn * HD_, 128, 32, HD_, HD_, 5, 3);
      tdm_load_tile(offV[buf ^ 1], Vbase + kn, 32, 128, L_, L_, 3, 3);
    }

    if (kb <= q0 + 15) {                  // causal: this wave has work here
      const char* Kt = &ldsK[buf][0];
      const char* Vl = &ldsV[buf][0];

      v8f s0 = (v8f){0.f,0.f,0.f,0.f,0.f,0.f,0.f,0.f};
      v8f s1 = s0;
#pragma unroll
      for (int kk = 0; kk < 4; ++kk) {
        v16bf bk0 = lds_b_frag(Kt,                  KT_STRIDE, kk * 32, lane);
        v16bf bk1 = lds_b_frag(Kt + 16 * KT_STRIDE, KT_STRIDE, kk * 32, lane);
        s0 = wmma_bf16(qf[kk], bk0, s0);
        s1 = wmma_bf16(qf[kk], bk1, s1);
      }

#pragma unroll
      for (int v = 0; v < 8; ++v) {
        int row = q0 + v + 8 * half;
        float x0 = s0[v] * ATTN_SCALE;
        float x1 = s1[v] * ATTN_SCALE;
        if (kb + nl > row)      x0 = -__builtin_inff();
        if (kb + 16 + nl > row) x1 = -__builtin_inff();
        s0[v] = x0; s1[v] = x1;
      }

      float alpha[8];
#pragma unroll
      for (int v = 0; v < 8; ++v) {
        float t = fmaxf(s0[v], s1[v]);
#pragma unroll
        for (int off = 8; off >= 1; off >>= 1) t = fmaxf(t, __shfl_xor(t, off, 32));
        float mnew = fmaxf(mrow[v], t);
        alpha[v] = __expf(mrow[v] - mnew);
        mrow[v] = mnew;
        float p0 = __expf(s0[v] - mnew);
        float p1 = __expf(s1[v] - mnew);
        s0[v] = p0; s1[v] = p1;
        float r = p0 + p1;
#pragma unroll
        for (int off = 8; off >= 1; off >>= 1) r += __shfl_xor(r, off, 32);
        lrow[v] = lrow[v] * alpha[v] + r;
      }

      // C-layout P -> A-layout via per-wave LDS tile (same wave, no barrier)
#pragma unroll
      for (int v = 0; v < 8; ++v) {
        int row = v + 8 * half;
        Pl[row * 32 + nl]      = (bf16)s0[v];
        Pl[row * 32 + 16 + nl] = (bf16)s1[v];
      }
      v16bf pf;
      {
        const bf16* p = Pl + (size_t)nl * 32 + half * 8;
        union { uint4 u[2]; v16bf v; } f;
        f.u[0] = *(const uint4*)(p);
        f.u[1] = *(const uint4*)(p + 16);
        pf = f.v;
      }

#pragma unroll
      for (int t = 0; t < 8; ++t) {
        v16bf bv = lds_b_frag(Vl + t * 16 * VT_STRIDE, VT_STRIDE, 0, lane);
#pragma unroll
        for (int v = 0; v < 8; ++v) o[t][v] *= alpha[v];
        o[t] = wmma_bf16(pf, bv, o[t]);
      }
    }

    __builtin_amdgcn_s_wait_tensorcnt(0);   // wave0: next slab landed
    __syncthreads();                        // all waves done with buf
  }

  size_t ybase = (size_t)b * L_ * D_;
#pragma unroll
  for (int t = 0; t < 8; ++t)
#pragma unroll
    for (int v = 0; v < 8; ++v) {
      int row = q0 + v + 8 * half;
      float val = o[t][v] / lrow[v];
      Yb[ybase + (size_t)row * D_ + h * HD_ + t * 16 + nl] = (bf16)val;
    }
}

// ---------------------------------------------------------------------------
// Host-side launcher
// ---------------------------------------------------------------------------
extern "C" void kernel_launch(void* const* d_in, const int* in_sizes, int n_in,
                              void* d_out, int out_size, void* d_ws, size_t ws_size,
                              hipStream_t stream) {
  (void)in_sizes; (void)n_in; (void)out_size; (void)ws_size;
  const float* x     = (const float*)d_in[0];   // (B,L,D)
  const float* w_qkv = (const float*)d_in[1];   // (3D,D)
  const float* w_o   = (const float*)d_in[2];   // (D,D)
  float* out = (float*)d_out;                   // (B,L,D) fp32

  const size_t nX    = (size_t)B_ * L_ * D_;
  const size_t nWqkv = (size_t)3 * D_ * D_;
  const size_t nWo   = (size_t)D_ * D_;

  char* ws = (char*)d_ws;
  bf16*  xb    = (bf16*)ws;   ws += nX * sizeof(bf16);
  bf16*  wqb   = (bf16*)ws;   ws += nWqkv * sizeof(bf16);
  bf16*  wob   = (bf16*)ws;   ws += nWo * sizeof(bf16);
  float* qkv32 = (float*)ws;  ws += (size_t)B_ * L_ * 3 * D_ * sizeof(float);
  bf16*  Qb    = (bf16*)ws;   ws += nX * sizeof(bf16);
  bf16*  Kb    = (bf16*)ws;   ws += nX * sizeof(bf16);
  bf16*  Vt    = (bf16*)ws;   ws += nX * sizeof(bf16);
  bf16*  Yb    = (bf16*)ws;   ws += nX * sizeof(bf16);

  cvt_f32_bf16<<<dim3((nX + 255) / 256), dim3(256), 0, stream>>>(x, xb, nX);
  cvt_f32_bf16<<<dim3((nWqkv + 255) / 256), dim3(256), 0, stream>>>(w_qkv, wqb, nWqkv);
  cvt_f32_bf16<<<dim3((nWo + 255) / 256), dim3(256), 0, stream>>>(w_o, wob, nWo);

  // QKV projection: (8192 x 2048) * (6144 x 2048)^T
  gemm_tdm_kernel<3 * D_, D_><<<dim3(3 * D_ / 128, (B_ * L_) / 128),
                                dim3(256), 0, stream>>>(xb, wqb, qkv32);

  {
    size_t total = (size_t)B_ * L_ * H_ * (HD_ / 2);
    rope_split_kernel<<<dim3((total + 255) / 256), dim3(256), 0, stream>>>(
        qkv32, Qb, Kb, Vt);
  }

  // attention: one block = 4 waves on 64 queries of one (b,h)
  attn_kernel<<<dim3(B_ * H_ * (L_ / 64)), dim3(128), 0, stream>>>(Qb, Kb, Vt, Yb);

  // Output projection: (8192 x 2048) * (2048 x 2048)^T
  gemm_tdm_kernel<D_, D_><<<dim3(D_ / 128, (B_ * L_) / 128),
                            dim3(256), 0, stream>>>(Yb, wob, out);
}
